// MinGPTModel_76605036692065
// MI455X (gfx1250) — matrix operand order repaired and checked by reading, not compiled, not run on hardware
//
#include <hip/hip_runtime.h>
#include <hip/hip_bf16.h>

// ---------------------------------------------------------------------------
// minGPT forward for MI455X (gfx1250, wave32, WMMA, async LDS copies).
// Weights pre-converted to f16 and pre-transposed to [N,K] once per launch.
// GEMM-feeding activations kept in f16; residual stream in f32.
// ---------------------------------------------------------------------------

typedef __attribute__((ext_vector_type(16))) _Float16 v16h;
typedef __attribute__((ext_vector_type(8)))  _Float16 v8h;
typedef __attribute__((ext_vector_type(8)))  float    v8f;

#define GB 4
#define GS 1024
#define GD 1024
#define GH 16
#define GHS 64
#define GDFF 4096
#define GV 32000
#define GL 4
#define GMS (GB*GS)
#define GSCALE 0.03125f    // D**-0.5

// ---- CDNA5 async global->LDS copy (ASYNCcnt) with safe fallback -----------
#if defined(__has_builtin)
#if __has_builtin(__builtin_amdgcn_global_load_async_to_lds_b128)
#define HAVE_ASYNC 1
#endif
#endif
#ifndef HAVE_ASYNC
#define HAVE_ASYNC 0
#endif

#if HAVE_ASYNC
// ROCm 7.2 prototype (from diagnostic): (v4i as1* src, v4i as3* lds, Imm, Imm)
typedef __attribute__((__vector_size__(4 * sizeof(int)))) int v4i_raw;
typedef __attribute__((address_space(1))) v4i_raw* gv4i_p;
typedef __attribute__((address_space(3))) v4i_raw* lv4i_p;
#endif

// copy 16 bytes from global to LDS (per lane)
__device__ inline void cp16(void* lds, const void* g) {
#if HAVE_ASYNC
    __builtin_amdgcn_global_load_async_to_lds_b128((gv4i_p)g, (lv4i_p)lds, 0, 0);
#else
    *(int4*)lds = *(const int4*)g;
#endif
}

__device__ inline void wait_async() {
#if HAVE_ASYNC
#if __has_builtin(__builtin_amdgcn_s_wait_asynccnt)
    __builtin_amdgcn_s_wait_asynccnt(0);
#else
    asm volatile("s_wait_asynccnt 0x0" ::: "memory");
#endif
#endif
}

// ---- WMMA fragment loaders (from 16B-aligned LDS rows) --------------------
// A fragment (16x32 f16): lane holds row m=l&15.
__device__ inline v16h load_a_frag(const _Float16* rowPtr, int half, int koff) {
    const v8h lo = *(const v8h*)(rowPtr + koff + half * 8);
    const v8h hi = *(const v8h*)(rowPtr + koff + 16 + half * 8);
    v16h a;
#pragma unroll
    for (int i = 0; i < 8; ++i) { a[i] = lo[i]; a[i + 8] = hi[i]; }
    return a;
}

// B fragment (32x16 f16) from TRANSPOSED LDS tile Bt[n][k].
__device__ inline v16h load_b_frag(const _Float16* rowPtr, int half) {
    const v8h lo = *(const v8h*)(rowPtr + half * 8);
    const v8h hi = *(const v8h*)(rowPtr + 16 + half * 8);
    v16h b;
#pragma unroll
    for (int i = 0; i < 8; ++i) { b[i] = lo[i]; b[i + 8] = hi[i]; }
    return b;
}

__device__ inline v8f wmma_f16(v16h a, v16h b, v8f c) {
    return __builtin_amdgcn_wmma_f32_16x16x32_f16(false, a, false, b,
                                                  (short)0, c, false, false);
}

// ---- Weight convert + transpose: Wt[n][k] = (f16)W[k][n] ------------------
__global__ __launch_bounds__(256) void mingpt_wcvt(
    const float* __restrict__ W, _Float16* __restrict__ Wt, int K, int N) {
    __shared__ float tile[32][33];
    const int t = threadIdx.x, tx = t & 31, ty = t >> 5;     // ty 0..7
    const int n0 = blockIdx.x * 32, k0 = blockIdx.y * 32;
#pragma unroll
    for (int i = 0; i < 4; ++i)
        tile[ty + i * 8][tx] = W[(size_t)(k0 + ty + i * 8) * N + n0 + tx];
    __syncthreads();
#pragma unroll
    for (int i = 0; i < 4; ++i)
        Wt[(size_t)(n0 + ty + i * 8) * K + k0 + tx] = (_Float16)tile[tx][ty + i * 8];
}

// ---- Embedding ------------------------------------------------------------
__global__ __launch_bounds__(256) void mingpt_embed(
    const int* __restrict__ idx, const float* __restrict__ tok,
    const float* __restrict__ pos, float* __restrict__ x) {
    const size_t row = blockIdx.x;
    const int tokid = idx[row];
    const int s = (int)(row & (GS - 1));
    const int t = threadIdx.x;
    const float4 tv = ((const float4*)(tok + (size_t)tokid * GD))[t];
    const float4 pv = ((const float4*)(pos + (size_t)s * GD))[t];
    float4 o; o.x = tv.x + pv.x; o.y = tv.y + pv.y; o.z = tv.z + pv.z; o.w = tv.w + pv.w;
    ((float4*)(x + row * GD))[t] = o;
}

// ---- LayerNorm over D=1024; writes f32 (residual) + f16 (GEMM input) ------
__global__ __launch_bounds__(256) void mingpt_ln(
    const float* __restrict__ x, const float* __restrict__ g,
    const float* __restrict__ bb, float* __restrict__ y32,
    _Float16* __restrict__ y16) {
    __shared__ float red[256];
    const int t = threadIdx.x;
    const size_t row = blockIdx.x;
    const float4 v = ((const float4*)(x + row * GD))[t];
    red[t] = v.x + v.y + v.z + v.w;
    __syncthreads();
#pragma unroll
    for (int o = 128; o > 0; o >>= 1) { if (t < o) red[t] += red[t + o]; __syncthreads(); }
    const float mu = red[0] * (1.0f / GD);
    __syncthreads();
    const float dx = v.x - mu, dy = v.y - mu, dz = v.z - mu, dw = v.w - mu;
    red[t] = dx * dx + dy * dy + dz * dz + dw * dw;
    __syncthreads();
#pragma unroll
    for (int o = 128; o > 0; o >>= 1) { if (t < o) red[t] += red[t + o]; __syncthreads(); }
    const float rstd = rsqrtf(red[0] * (1.0f / GD) + 1e-5f);
    const float4 gv = ((const float4*)g)[t];
    const float4 bv = ((const float4*)bb)[t];
    float4 o4;
    o4.x = dx * rstd * gv.x + bv.x;
    o4.y = dy * rstd * gv.y + bv.y;
    o4.z = dz * rstd * gv.z + bv.z;
    o4.w = dw * rstd * gv.w + bv.w;
    ((float4*)(y32 + row * GD))[t] = o4;
    _Float16* d = y16 + row * GD + (size_t)t * 4;
    d[0] = (_Float16)o4.x; d[1] = (_Float16)o4.y;
    d[2] = (_Float16)o4.z; d[3] = (_Float16)o4.w;
}

// ---- WMMA GEMM: C[M,N] = act(A[M,K](f16) @ Wt[N,K](f16) + bias + resid) ---
// Block tile 128x128, 8 waves, each wave 32x64 (2x4 of 16x16x32 WMMA).
template <int BIAS, int RESID, int RELU, int OUT16>
__global__ __launch_bounds__(256) void mingpt_gemm(
    const _Float16* __restrict__ A, const _Float16* __restrict__ Wt,
    const float* __restrict__ bias, const float* __restrict__ resid,
    float* __restrict__ C32, _Float16* __restrict__ C16,
    int M, int N, int K) {
    __shared__ __align__(16) _Float16 As[128][40];   // 80B row stride
    __shared__ __align__(16) _Float16 Bt[128][40];

    const int t = threadIdx.x;
    const int lane = t & 31, wv = t >> 5;
    const int wm = wv & 3, wn = wv >> 2;
    const int half = lane >> 4, l16 = lane & 15;
    const int m0 = blockIdx.y * 128, n0 = blockIdx.x * 128;

    v8f acc[2][4] = {};

    const int arow = t >> 1, ac = (t & 1) * 16;      // 2x16B per thread per tile

    for (int k0 = 0; k0 < K; k0 += 32) {
        __syncthreads();   // WAR vs previous fragment reads
        const _Float16* ap = A  + (size_t)(m0 + arow) * K + k0 + ac;
        const _Float16* wp = Wt + (size_t)(n0 + arow) * K + k0 + ac;
        cp16(&As[arow][ac],     ap);
        cp16(&As[arow][ac + 8], ap + 8);
        cp16(&Bt[arow][ac],     wp);
        cp16(&Bt[arow][ac + 8], wp + 8);
        if (k0 + 32 < K) {
            __builtin_prefetch(ap + 32, 0, 1);
            __builtin_prefetch(wp + 32, 0, 1);
        }
        wait_async();
        __syncthreads();
        const v16h a0 = load_a_frag(&As[wm * 32      + l16][0], half, 0);
        const v16h a1 = load_a_frag(&As[wm * 32 + 16 + l16][0], half, 0);
#pragma unroll
        for (int j = 0; j < 4; ++j) {
            const v16h bf = load_b_frag(&Bt[wn * 64 + j * 16 + l16][0], half);
            acc[0][j] = wmma_f16(a0, bf, acc[0][j]);
            acc[1][j] = wmma_f16(a1, bf, acc[1][j]);
        }
    }

#pragma unroll
    for (int i = 0; i < 2; ++i) {
#pragma unroll
        for (int j = 0; j < 4; ++j) {
            const int rowb = m0 + wm * 32 + i * 16 + half * 8;
            const int col  = n0 + wn * 64 + j * 16 + l16;
            const float badd = BIAS ? bias[col] : 0.0f;
#pragma unroll
            for (int r = 0; r < 8; ++r) {
                float v = acc[i][j][r] + badd;
                if (RESID) v += resid[(size_t)(rowb + r) * N + col];
                if (RELU)  v = fmaxf(v, 0.0f);
                if (OUT16) C16[(size_t)(rowb + r) * N + col] = (_Float16)v;
                else       C32[(size_t)(rowb + r) * N + col] = v;
            }
        }
    }
}

// ---- Flash attention (causal): per block one (b, h, 64-query tile) --------
__global__ __launch_bounds__(128) void mingpt_attn(
    const _Float16* __restrict__ qb, const _Float16* __restrict__ kb,
    const _Float16* __restrict__ vb, _Float16* __restrict__ ob) {
    __shared__ __align__(16) _Float16 Qs[64][72];      // [qrow][hs] (unscaled)
    __shared__ __align__(16) _Float16 Ks[32][72];      // [key][hs]
    __shared__ __align__(16) _Float16 Vt[64][40];      // [hs][key]
    __shared__ __align__(16) _Float16 Ps[4][16][40];   // per-wave P tile

    const int t = threadIdx.x;
    const int lane = t & 31, w = t >> 5;
    const int half = lane >> 4, l16 = lane & 15;
    const int qt = blockIdx.x & 15;
    const int h  = (blockIdx.x >> 4) & 15;
    const int b  = blockIdx.x >> 8;
    const int q0 = qt * 64;
    const size_t baseBS = (size_t)b * GS;

    // stage Q (async 16B copies; scale folded into softmax)
    {
        const int row = t >> 1, c0 = (t & 1) * 32;
        const _Float16* src = qb + (baseBS + q0 + row) * GD + h * GHS + c0;
#pragma unroll
        for (int i = 0; i < 32; i += 8) cp16(&Qs[row][c0 + i], src + i);
    }
    wait_async();
    __syncthreads();
    const v16h aQ0 = load_a_frag(&Qs[w * 16 + l16][0], half, 0);
    const v16h aQ1 = load_a_frag(&Qs[w * 16 + l16][0], half, 32);

    v8f o0 = {}, o1 = {}, o2 = {}, o3 = {};
    float mrow[8], srow[8];
#pragma unroll
    for (int r = 0; r < 8; ++r) { mrow[r] = -1e30f; srow[r] = 0.0f; }
    const int rbase = q0 + w * 16 + half * 8;
    const int nkt = (q0 + 64) >> 5;

    for (int kt = 0; kt < nkt; ++kt) {
        const int kk0 = kt * 32;
        __syncthreads();   // WAR vs previous P.V reads
        {   // K tile [32][64], async
            const int key = t >> 2, c0 = (t & 3) * 16;
            const _Float16* src = kb + (baseBS + kk0 + key) * GD + h * GHS + c0;
            cp16(&Ks[key][c0], src);
            cp16(&Ks[key][c0 + 8], src + 8);
        }
        {   // V tile transposed [64][32]
            const int hs = t & 63, kv0 = (t >> 6) * 16;
            const _Float16* src = vb + (baseBS + kk0 + kv0) * GD + h * GHS + hs;
#pragma unroll
            for (int i = 0; i < 16; ++i) Vt[hs][kv0 + i] = src[(size_t)i * GD];
        }
        wait_async();
        __syncthreads();

        // scores: 16x32 per wave (contraction over HS=64)
        v8f s0 = {}, s1 = {};
        {
            const v16h b00 = load_b_frag(&Ks[l16][0],       half);
            const v16h b01 = load_b_frag(&Ks[l16][32],      half);
            const v16h b10 = load_b_frag(&Ks[16 + l16][0],  half);
            const v16h b11 = load_b_frag(&Ks[16 + l16][32], half);
            s0 = wmma_f16(aQ0, b00, s0); s0 = wmma_f16(aQ1, b01, s0);
            s1 = wmma_f16(aQ0, b10, s1); s1 = wmma_f16(aQ1, b11, s1);
        }

        // causal mask + online softmax
        const int c0 = kk0 + l16;
#pragma unroll
        for (int r = 0; r < 8; ++r) {
            const int row = rbase + r;
            const float v0 = (c0      > row) ? -1e30f : s0[r] * GSCALE;
            const float v1 = (c0 + 16 > row) ? -1e30f : s1[r] * GSCALE;
            float mx = fmaxf(v0, v1);
            mx = fmaxf(mx, __shfl_xor(mx, 1));
            mx = fmaxf(mx, __shfl_xor(mx, 2));
            mx = fmaxf(mx, __shfl_xor(mx, 4));
            mx = fmaxf(mx, __shfl_xor(mx, 8));
            const float mnew = fmaxf(mrow[r], mx);
            const float corr = __expf(mrow[r] - mnew);
            const float p0 = __expf(v0 - mnew);
            const float p1 = __expf(v1 - mnew);
            float rs = p0 + p1;
            rs += __shfl_xor(rs, 1);
            rs += __shfl_xor(rs, 2);
            rs += __shfl_xor(rs, 4);
            rs += __shfl_xor(rs, 8);
            srow[r] = srow[r] * corr + rs;
            mrow[r] = mnew;
            o0[r] *= corr; o1[r] *= corr; o2[r] *= corr; o3[r] *= corr;
            Ps[w][half * 8 + r][l16]      = (_Float16)p0;
            Ps[w][half * 8 + r][16 + l16] = (_Float16)p1;
        }
        __syncthreads();

        // out += P(16x32) @ V(32x64)
        const v16h aP  = load_a_frag(&Ps[w][l16][0], half, 0);
        const v16h bv0 = load_b_frag(&Vt[ 0 + l16][0], half);
        const v16h bv1 = load_b_frag(&Vt[16 + l16][0], half);
        const v16h bv2 = load_b_frag(&Vt[32 + l16][0], half);
        const v16h bv3 = load_b_frag(&Vt[48 + l16][0], half);
        o0 = wmma_f16(aP, bv0, o0);
        o1 = wmma_f16(aP, bv1, o1);
        o2 = wmma_f16(aP, bv2, o2);
        o3 = wmma_f16(aP, bv3, o3);
    }

#pragma unroll
    for (int r = 0; r < 8; ++r) {
        const float inv = 1.0f / srow[r];
        _Float16* dst = ob + (baseBS + rbase + r) * GD + h * GHS;
        dst[ 0 + l16] = (_Float16)(o0[r] * inv);
        dst[16 + l16] = (_Float16)(o1[r] * inv);
        dst[32 + l16] = (_Float16)(o2[r] * inv);
        dst[48 + l16] = (_Float16)(o3[r] * inv);
    }
}

// ---------------------------------------------------------------------------
extern "C" void kernel_launch(void* const* d_in, const int* in_sizes, int n_in,
                              void* d_out, int out_size, void* d_ws, size_t ws_size,
                              hipStream_t stream) {
    (void)in_sizes; (void)n_in; (void)out_size; (void)ws_size;
    const int*   idx    = (const int*)  d_in[0];
    const float* tok    = (const float*)d_in[1];
    const float* pos    = (const float*)d_in[2];
    const float* ln1_g  = (const float*)d_in[3];
    const float* ln1_b  = (const float*)d_in[4];
    const float* ln2_g  = (const float*)d_in[5];
    const float* ln2_b  = (const float*)d_in[6];
    const float* wq     = (const float*)d_in[7];
    const float* wk     = (const float*)d_in[8];
    const float* wv     = (const float*)d_in[9];
    const float* wo     = (const float*)d_in[10];
    const float* bo     = (const float*)d_in[11];
    const float* w1     = (const float*)d_in[12];
    const float* b1     = (const float*)d_in[13];
    const float* w2     = (const float*)d_in[14];
    const float* b2     = (const float*)d_in[15];
    const float* lnf_g  = (const float*)d_in[16];
    const float* lnf_b  = (const float*)d_in[17];
    const float* w_head = (const float*)d_in[18];
    const float* b_head = (const float*)d_in[19];
    float* out = (float*)d_out;

    // ---- workspace layout ----
    const size_t RD = (size_t)GMS * GD;        // 4M
    const size_t RF = (size_t)GMS * GDFF;      // 16M
    const size_t DD = (size_t)GD * GD;         // 1M
    const size_t DF = (size_t)GD * GDFF;       // 4M
    char* p = (char*)d_ws;
    float*    x    = (float*)p;    p += RD * 4;
    float*    xn32 = (float*)p;    p += RD * 4;
    _Float16* xnh  = (_Float16*)p; p += RD * 2;
    _Float16* qh   = (_Float16*)p; p += RD * 2;
    _Float16* kh   = (_Float16*)p; p += RD * 2;
    _Float16* vh   = (_Float16*)p; p += RD * 2;
    _Float16* th   = (_Float16*)p; p += RD * 2;
    _Float16* hh   = (_Float16*)p; p += RF * 2;
    _Float16* wtq  = (_Float16*)p; p += GL * DD * 2;
    _Float16* wtk  = (_Float16*)p; p += GL * DD * 2;
    _Float16* wtv  = (_Float16*)p; p += GL * DD * 2;
    _Float16* wto  = (_Float16*)p; p += GL * DD * 2;
    _Float16* wt1  = (_Float16*)p; p += GL * DF * 2;
    _Float16* wt2  = (_Float16*)p; p += GL * DF * 2;
    _Float16* wth  = (_Float16*)p; p += (size_t)GD * GV * 2;

    const dim3 blk256(256), blk128(128);
    const dim3 gRows(GMS);
    const dim3 gGemmD(GD / 128, GMS / 128);       // N=1024
    const dim3 gGemmF(GDFF / 128, GMS / 128);     // N=4096
    const dim3 gGemmV(GV / 128, GMS / 128);       // N=32000
    const dim3 gAttn(GB * GH * (GS / 64));
    const dim3 gCvtDD(GD / 32, GD / 32);
    const dim3 gCvtDF(GDFF / 32, GD / 32);        // w1: K=GD, N=GDFF
    const dim3 gCvtFD(GD / 32, GDFF / 32);        // w2: K=GDFF, N=GD
    const dim3 gCvtDV(GV / 32, GD / 32);

    // ---- pre-convert + transpose all weights to f16 [N,K] ----
    for (int l = 0; l < GL; ++l) {
        mingpt_wcvt<<<gCvtDD, blk256, 0, stream>>>(wq + (size_t)l * DD, wtq + (size_t)l * DD, GD, GD);
        mingpt_wcvt<<<gCvtDD, blk256, 0, stream>>>(wk + (size_t)l * DD, wtk + (size_t)l * DD, GD, GD);
        mingpt_wcvt<<<gCvtDD, blk256, 0, stream>>>(wv + (size_t)l * DD, wtv + (size_t)l * DD, GD, GD);
        mingpt_wcvt<<<gCvtDD, blk256, 0, stream>>>(wo + (size_t)l * DD, wto + (size_t)l * DD, GD, GD);
        mingpt_wcvt<<<gCvtDF, blk256, 0, stream>>>(w1 + (size_t)l * DF, wt1 + (size_t)l * DF, GD, GDFF);
        mingpt_wcvt<<<gCvtFD, blk256, 0, stream>>>(w2 + (size_t)l * DF, wt2 + (size_t)l * DF, GDFF, GD);
    }
    mingpt_wcvt<<<gCvtDV, blk256, 0, stream>>>(w_head, wth, GD, GV);

    mingpt_embed<<<gRows, blk256, 0, stream>>>(idx, tok, pos, x);

    for (int l = 0; l < GL; ++l) {
        const size_t dd = (size_t)l * DD;
        const size_t df = (size_t)l * DF;
        // xn = LN1(x)
        mingpt_ln<<<gRows, blk256, 0, stream>>>(x, ln1_g + l * GD, ln1_b + l * GD, xn32, xnh);
        // q,k,v = xn @ Wq/Wk/Wv  (f16 out)
        mingpt_gemm<0,0,0,1><<<gGemmD, blk256, 0, stream>>>(xnh, wtq + dd, nullptr, nullptr, nullptr, qh, GMS, GD, GD);
        mingpt_gemm<0,0,0,1><<<gGemmD, blk256, 0, stream>>>(xnh, wtk + dd, nullptr, nullptr, nullptr, kh, GMS, GD, GD);
        mingpt_gemm<0,0,0,1><<<gGemmD, blk256, 0, stream>>>(xnh, wtv + dd, nullptr, nullptr, nullptr, vh, GMS, GD, GD);
        // th = causal_attention(q,k,v)
        mingpt_attn<<<gAttn, blk128, 0, stream>>>(qh, kh, vh, th);
        // x = xn + th @ Wo + bo   (f32 out)
        mingpt_gemm<1,1,0,0><<<gGemmD, blk256, 0, stream>>>(th, wto + dd, bo + l * GD, xn32, x, nullptr, GMS, GD, GD);
        // xn = LN2(x)
        mingpt_ln<<<gRows, blk256, 0, stream>>>(x, ln2_g + l * GD, ln2_b + l * GD, xn32, xnh);
        // hh = relu(xn @ W1 + b1)  (f16 out)
        mingpt_gemm<1,0,1,1><<<gGemmF, blk256, 0, stream>>>(xnh, wt1 + df, b1 + (size_t)l * GDFF, nullptr, nullptr, hh, GMS, GDFF, GD);
        // x = xn + hh @ W2 + b2   (f32 out)
        mingpt_gemm<1,1,0,0><<<gGemmD, blk256, 0, stream>>>(hh, wt2 + df, b2 + l * GD, xn32, x, nullptr, GMS, GD, GDFF);
    }
    // final LN + head
    mingpt_ln<<<gRows, blk256, 0, stream>>>(x, lnf_g, lnf_b, xn32, xnh);
    mingpt_gemm<1,0,0,0><<<gGemmV, blk256, 0, stream>>>(xnh, wth, b_head, nullptr, out, nullptr, GMS, GV, GD);
}
